// LNN_1769526526451
// MI455X (gfx1250) — compile-verified, matrix operand order checked by reading, not run on hardware
//
#include <hip/hip_runtime.h>
#include <stdint.h>

// ---------------------------------------------------------------------------
// CfC (liquid NN) scan on MI455X (gfx1250, wave32, WMMA bf16).
// Hidden-split cooperative design:
//   * 16 workgroups process ALL 128 batch rows; WG g owns hidden cols
//     [32g, 32g+32) of H=512 (=> 128 of the 2048 LSTM gate columns).
//   * Weight slices (Wh 128KB + Wi 16KB + ff 32KB) are LDS-resident,
//     staged once (Wh, ff via the Tensor Data Mover; Wi by a cooperative copy).
//   * Per-step cross-WG activation exchange (h, h_lstm, f) through L2
//     with 3 atomic grid syncs per step -> per-WGP L2 traffic drops from
//     2.9MB/step (batch-split) to ~0.3MB/step; compute/memory balanced.
//   * All GEMMs: v_wmma_f32_16x16x32_bf16, f32 accumulation; cell state c
//     in f32 registers; head fused with shfl_xor reductions.
// ---------------------------------------------------------------------------

typedef __attribute__((ext_vector_type(16))) __bf16 v16bf;
typedef __attribute__((ext_vector_type(8)))  float  v8f;
typedef __attribute__((ext_vector_type(4)))  unsigned int v4u;
typedef __attribute__((ext_vector_type(8)))  int v8i;
typedef __attribute__((ext_vector_type(4)))  int v4i;

#define B_   128
#define T_   1024
#define I_   64
#define H_   512
#define BB_  128
#define O_   2
#define G4H  2048
#define NWG  16
// dynamic LDS: sWh 128KB | sWi 16KB | sFF 32KB
#define LDS_WH_OFF  0
#define LDS_WI_OFF  131072
#define LDS_FF_OFF  147456
#define DYN_LDS     180224

static __device__ __forceinline__ __bf16 f2bf(float f) {
  uint32_t u = __builtin_bit_cast(uint32_t, f);
  uint32_t r = u + 0x7FFFu + ((u >> 16) & 1u);   // round-to-nearest-even
  uint16_t h = (uint16_t)(r >> 16);
  return __builtin_bit_cast(__bf16, h);
}
static __device__ __forceinline__ float bf2f(__bf16 b) {
  uint16_t h = __builtin_bit_cast(uint16_t, b);
  uint32_t u = ((uint32_t)h) << 16;
  return __builtin_bit_cast(float, u);
}
static __device__ __forceinline__ uint32_t pk2(float a, float b) {
  return (uint32_t)__builtin_bit_cast(uint16_t, f2bf(a)) |
         ((uint32_t)__builtin_bit_cast(uint16_t, f2bf(b)) << 16);
}
static __device__ __forceinline__ float sigm(float x) {
  return __builtin_amdgcn_rcpf(1.0f + __expf(-x));
}
static __device__ __forceinline__ float tanh_f(float x) {
  return 2.0f * sigm(2.0f * x) - 1.0f;
}

// ---------------------------------------------------------------------------
__global__ void cvt_bf16x4(const float4* __restrict__ src, uint2* __restrict__ dst, int n4) {
  int i = blockIdx.x * blockDim.x + threadIdx.x;
  if (i < n4) {
    float4 v = src[i];
    uint2 r;
    r.x = pk2(v.x, v.y);
    r.y = pk2(v.z, v.w);
    dst[i] = r;
  }
}

// ---------------------------------------------------------------------------
// TDM 2D tile load: global (row-major, 8-byte units) -> LDS.
// D# packed per CDNA5 ISA 8.3/8.4 (group0: count/lds_addr/global_addr/type,
// group1: data_size=8B, dims/strides). 2D => remaining groups zero.
// clang-23 toolchain exposes the 6-arg form:
//   (uint32x4 g0, int32x8 g1, int32x4, int32x4, int32x8, i32 cpol)
static __device__ __forceinline__ uint32_t lds_off(const void* p) {
  return (uint32_t)(uintptr_t)p;   // low 32 bits of flat shared addr = LDS offset
}

static __device__ __forceinline__ void tdm_load_2d(uint32_t lds_byte, const void* gptr,
                                                   uint32_t width_u,   // tile/tensor dim0 (8B units)
                                                   uint32_t rows,      // tile dim1
                                                   uint32_t stride_u)  // row stride (8B units)
{
#if __has_builtin(__builtin_amdgcn_tensor_load_to_lds)
  const uint64_t ga = (uint64_t)gptr;
  v4u g0;
  g0.x = 1u;                                      // count=1 user descriptor
  g0.y = lds_byte;                                // lds_addr [63:32]
  g0.z = (uint32_t)(ga & 0xffffffffu);            // global_addr[31:0]  -> bits 95:64
  g0.w = (uint32_t)((ga >> 32) & 0x1ffffffu) |    // global_addr[56:32] -> bits 120:96
         (2u << 30);                              // type=2 ("image")   -> bits 127:126
  v8i g1;
  g1[0] = (int)(3u << 16);                        // workgroup_mask=0, data_size=3 (8B)
  g1[1] = (int)((width_u & 0xffffu) << 16);       // tensor_dim0[15:0] -> bits 63:48
  g1[2] = (int)(((width_u >> 16) & 0xffffu) |     // tensor_dim0[31:16]
                ((rows & 0xffffu) << 16));        // tensor_dim1[15:0]
  g1[3] = (int)(((rows >> 16) & 0xffffu) |        // tensor_dim1[31:16]
                ((width_u & 0xffffu) << 16));     // tile_dim0
  g1[4] = (int)(rows & 0xffffu);                  // tile_dim1 (tile_dim2 = 0)
  g1[5] = (int)stride_u;                          // tensor_dim0_stride[31:0]
  g1[6] = 0;                                      // stride[47:32] | dim1_stride lo
  g1[7] = 0;
  v4i gz4 = {0, 0, 0, 0};
  v8i gz8 = {0, 0, 0, 0, 0, 0, 0, 0};
  __builtin_amdgcn_tensor_load_to_lds(g0, g1, gz4, gz4, gz8, 0);
#else
  (void)lds_byte; (void)gptr; (void)width_u; (void)rows; (void)stride_u;
#endif
}

// ---------------------------------------------------------------------------
// Monotonic-counter grid barrier (16 WGs, counter reset each launch).
static __device__ __forceinline__ void grid_sync(unsigned* cnt, unsigned target) {
  __threadfence();                 // flush this thread's global stores to L2
  __syncthreads();
  if (threadIdx.x == 0) {
    __hip_atomic_fetch_add(cnt, 1u, __ATOMIC_RELEASE, __HIP_MEMORY_SCOPE_AGENT);
    while (__hip_atomic_load(cnt, __ATOMIC_ACQUIRE, __HIP_MEMORY_SCOPE_AGENT) < target) {
      __builtin_amdgcn_s_sleep(1);
    }
  }
  __syncthreads();
  __threadfence();                 // acquire: drop stale near-cache lines
}

// ---------------------------------------------------------------------------
__global__ __launch_bounds__(256)
void cfc_scan_kernel(const __bf16* __restrict__ xbf,      // (B,T,I) bf16
                     const float*  __restrict__ lstm_bi,  // (4H)
                     const float*  __restrict__ bb_b,     // (BB)
                     const float*  __restrict__ ff1_b,
                     const float*  __restrict__ ff2_b,
                     const float*  __restrict__ ta_b,
                     const float*  __restrict__ tb_b,
                     const float*  __restrict__ head_W,   // (O,H)
                     const float*  __restrict__ head_b,   // (O)
                     const __bf16* __restrict__ Wi,       // (4H,I)   bf16
                     const __bf16* __restrict__ Wh,       // (4H,H)   bf16
                     const __bf16* __restrict__ Wbb,      // (BB,I+H) bf16
                     const __bf16* __restrict__ Wff,      // 4 x (H,BB) bf16
                     __bf16* __restrict__ hx,             // (B,H)  h carry exchange
                     __bf16* __restrict__ hlx,            // (B,H)  h_lstm exchange
                     __bf16* __restrict__ fx,             // (B,BB) backbone exchange
                     float*  __restrict__ hpart,          // (NWG,B,O) head partials
                     unsigned* __restrict__ sync,         // grid barrier counter
                     float*  __restrict__ out)            // phys(B,T,O) ++ last_h(B,H)
{
  extern __shared__ __align__(16) char smem[];
  __bf16* sWh = (__bf16*)(smem + LDS_WH_OFF);   // [128][512]: rows g4*32 + r
  __bf16* sWi = (__bf16*)(smem + LDS_WI_OFF);   // [128][64]
  __bf16* sFF = (__bf16*)(smem + LDS_FF_OFF);   // [4][32][128]

  __shared__ float sBi[128];      // gate bias slice [g4*32 + c]
  __shared__ float sFb[128];      // ff bias slice   [mat*32 + c]
  __shared__ float sBb[16];       // bb bias slice
  __shared__ float sHW[64];       // head_W slice [o*32 + c]
  __shared__ float sHb[2];

  const int tid  = threadIdx.x;
  const int wid  = tid >> 5;
  const int lane = tid & 31;
  const int lm   = lane & 15;
  const int lh   = lane >> 4;
  const int wg   = blockIdx.x;    // 0..15, owns hidden cols [32wg, 32wg+32)

  // ---- stage weight slices ----
  {  // sWi manual cooperative copy (16KB)
    uint4* dW = (uint4*)sWi;
    for (int i = tid; i < (128 * I_) / 8; i += 256) {
      const int rr = i >> 3, kq = i & 7;
      const int g4 = rr >> 5, r = rr & 31;
      dW[i] = *(const uint4*)&Wi[(size_t)(g4 * H_ + 32 * wg + r) * I_ + kq * 8];
    }
  }
  if (tid < 128) sBi[tid] = lstm_bi[(tid >> 5) * H_ + 32 * wg + (tid & 31)];
  if (tid < 32) {
    sFb[tid]      = ff1_b[32 * wg + tid];
    sFb[32 + tid] = ff2_b[32 * wg + tid];
    sFb[64 + tid] = ta_b [32 * wg + tid];
    sFb[96 + tid] = tb_b [32 * wg + tid];
  }
  if (tid < 16) sBb[tid] = bb_b[16 * (wg & 7) + tid];
  if (tid < 64) sHW[tid] = head_W[(tid >> 5) * H_ + 32 * wg + (tid & 31)];
  if (tid < 2)  sHb[tid] = head_b[tid];

  if (wid == 0) {
    // TDM: Wh slice, 4 blocks of 32 rows x 1024B (8B units: width 128, stride 128)
    for (int g4 = 0; g4 < 4; ++g4)
      tdm_load_2d(lds_off(sWh) + (uint32_t)g4 * 32 * 1024,
                  (const char*)Wh + (size_t)(g4 * H_ + 32 * wg) * H_ * 2,
                  128u, 32u, 128u);
    // TDM: ff slices, 4 mats of 32 rows x 256B (8B units: width 32, stride 32)
    for (int m = 0; m < 4; ++m)
      tdm_load_2d(lds_off(sFF) + (uint32_t)m * 32 * 256,
                  (const char*)Wff + (size_t)(m * H_ + 32 * wg) * BB_ * 2,
                  32u, 32u, 32u);
#if __has_builtin(__builtin_amdgcn_s_wait_tensorcnt)
    __builtin_amdgcn_s_wait_tensorcnt(0);
#endif
  }
  __syncthreads();

  // head weight hoist (per lane, constant over t)
  const float w00 = sHW[lm], w01 = sHW[16 + lm];
  const float w10 = sHW[32 + lm], w11 = sHW[48 + lm];

  const v8f vzero = {};
  float creg[2][8];
  #pragma unroll
  for (int jb = 0; jb < 2; ++jb)
    #pragma unroll
    for (int e = 0; e < 8; ++e) creg[jb][e] = 0.0f;

  for (int t = 0; t < T_; ++t) {
    // ================= Phase G: LSTM gates (M=128, N=128 slice, K=64+512) =================
    {
      v8f acc[8];                       // u = g4*2 + jb
      #pragma unroll
      for (int u = 0; u < 8; ++u) acc[u] = vzero;

      const size_t xrow = ((size_t)(16 * wid + lm) * T_ + t) * I_;
      #pragma unroll
      for (int ks = 0; ks < 2; ++ks) {
        const int k0 = ks * 32 + lh * 16;
        v16bf a = *(const v16bf*)&xbf[xrow + k0];
        #pragma unroll
        for (int u = 0; u < 8; ++u) {
          v16bf bm = *(const v16bf*)&sWi[(u * 16 + lm) * I_ + k0];
          acc[u] = __builtin_amdgcn_wmma_f32_16x16x32_bf16(false, a, false, bm, (short)0, acc[u], false, false);
        }
      }
      const size_t hrow = (size_t)(16 * wid + lm) * H_;
      #pragma unroll 4
      for (int ks = 0; ks < 16; ++ks) {
        const int k0 = ks * 32 + lh * 16;
        v16bf a = *(const v16bf*)&hx[hrow + k0];
        #pragma unroll
        for (int u = 0; u < 8; ++u) {
          v16bf bm = *(const v16bf*)&sWh[(u * 16 + lm) * H_ + k0];
          acc[u] = __builtin_amdgcn_wmma_f32_16x16x32_bf16(false, a, false, bm, (short)0, acc[u], false, false);
        }
      }
      #pragma unroll
      for (int jb = 0; jb < 2; ++jb) {
        const int cl = jb * 16 + lm;
        const int gcol = 32 * wg + cl;
        const float b_i = sBi[cl], b_g = sBi[32 + cl], b_f = sBi[64 + cl], b_o = sBi[96 + cl];
        #pragma unroll
        for (int e = 0; e < 8; ++e) {
          const float zi = acc[0 + jb][e] + b_i;
          const float zg = acc[2 + jb][e] + b_g;
          const float zf = acc[4 + jb][e] + b_f + 1.0f;   // forget bias +1
          const float zo = acc[6 + jb][e] + b_o;
          const float cn = creg[jb][e] * sigm(zf) + tanh_f(zi) * sigm(zg);
          creg[jb][e] = cn;
          const int row = 16 * wid + e + 8 * lh;
          hlx[(size_t)row * H_ + gcol] = f2bf(tanh_f(cn) * sigm(zo));
        }
      }
    }
    grid_sync(sync, 16u * (3u * t + 1u));

    // ================= Phase BB: backbone (64 tiles over 16 WGs x 4 waves) =================
    if (wid < 4) {
      const int nb = wg & 7;                 // N-tile (16 of BB=128 cols)
      const int mt = (wg >> 3) * 4 + wid;    // M-tile (16 batch rows)
      v8f acc = vzero;
      const size_t xrow = ((size_t)(16 * mt + lm) * T_ + t) * I_;
      #pragma unroll
      for (int ks = 0; ks < 2; ++ks) {
        const int k0 = ks * 32 + lh * 16;
        v16bf a = *(const v16bf*)&xbf[xrow + k0];
        v16bf bm = *(const v16bf*)&Wbb[(size_t)(nb * 16 + lm) * (I_ + H_) + k0];
        acc = __builtin_amdgcn_wmma_f32_16x16x32_bf16(false, a, false, bm, (short)0, acc, false, false);
      }
      const size_t hrow = (size_t)(16 * mt + lm) * H_;
      #pragma unroll 4
      for (int ks = 0; ks < 16; ++ks) {
        const int k0 = ks * 32 + lh * 16;
        v16bf a = *(const v16bf*)&hlx[hrow + k0];
        v16bf bm = *(const v16bf*)&Wbb[(size_t)(nb * 16 + lm) * (I_ + H_) + I_ + k0];
        acc = __builtin_amdgcn_wmma_f32_16x16x32_bf16(false, a, false, bm, (short)0, acc, false, false);
      }
      const float bbv = sBb[lm];
      #pragma unroll
      for (int e = 0; e < 8; ++e) {
        const int row = 16 * mt + e + 8 * lh;
        fx[(size_t)row * BB_ + nb * 16 + lm] = f2bf(1.7159f * tanh_f(0.666f * (acc[e] + bbv)));
      }
    }
    grid_sync(sync, 16u * (3u * t + 2u));

    // ================= Phase FF: ff1/ff2/ta/tb (K=128) + h_new + head partials =================
    {
      v8f fa[8];                        // u = mat*2 + jb
      #pragma unroll
      for (int u = 0; u < 8; ++u) fa[u] = vzero;
      const size_t frow = (size_t)(16 * wid + lm) * BB_;
      #pragma unroll
      for (int ks = 0; ks < 4; ++ks) {
        const int k0 = ks * 32 + lh * 16;
        v16bf a = *(const v16bf*)&fx[frow + k0];
        #pragma unroll
        for (int u = 0; u < 8; ++u) {
          v16bf bm = *(const v16bf*)&sFF[(u * 16 + lm) * BB_ + k0];
          fa[u] = __builtin_amdgcn_wmma_f32_16x16x32_bf16(false, a, false, bm, (short)0, fa[u], false, false);
        }
      }
      float hn0[8], hn1[8];
      #pragma unroll
      for (int jb = 0; jb < 2; ++jb) {
        const int cl = jb * 16 + lm;
        const int gcol = 32 * wg + cl;
        const float b1 = sFb[cl], b2 = sFb[32 + cl], b3 = sFb[64 + cl], b4 = sFb[96 + cl];
        #pragma unroll
        for (int e = 0; e < 8; ++e) {
          const float f1  = tanh_f(fa[0 + jb][e] + b1);
          const float f2v = tanh_f(fa[2 + jb][e] + b2);
          const float ti  = sigm((fa[4 + jb][e] + b3) + (fa[6 + jb][e] + b4));  // ts = 1.0
          const float hn  = f1 * (1.0f - ti) + ti * f2v;
          (jb ? hn1 : hn0)[e] = hn;
          hx[(size_t)(16 * wid + e + 8 * lh) * H_ + gcol] = f2bf(hn);
        }
      }
      // head partials: reduce over this WG's 32 hidden cols via shfl_xor
      #pragma unroll
      for (int e = 0; e < 8; ++e) {
        float p0 = hn0[e] * w00 + hn1[e] * w01;
        float p1 = hn0[e] * w10 + hn1[e] * w11;
        #pragma unroll
        for (int d = 8; d >= 1; d >>= 1) {
          p0 += __shfl_xor(p0, d);
          p1 += __shfl_xor(p1, d);
        }
        if (lm == 0) {
          const int row = 16 * wid + e + 8 * lh;
          hpart[((size_t)wg * B_ + row) * O_ + 0] = p0;
          hpart[((size_t)wg * B_ + row) * O_ + 1] = p1;
        }
      }
    }
    grid_sync(sync, 16u * (3u * t + 3u));

    // ================= Phase HD: final head sum (WG wg -> batch rows [8wg,8wg+8)) =================
    if (tid < 16) {
      const int row = 8 * wg + (tid >> 1);
      const int o = tid & 1;
      float s = sHb[o];
      #pragma unroll 4
      for (int g = 0; g < NWG; ++g) s += hpart[((size_t)g * B_ + row) * O_ + o];
      out[((size_t)row * T_ + t) * O_ + o] = s;
    }
  }

  // ---- last_h (f32), WG wg writes rows [8wg, 8wg+8) ----
  for (int i = tid; i < 8 * H_; i += 256) {
    const int r = 8 * wg + (i >> 9), c = i & (H_ - 1);
    out[(size_t)B_ * T_ * O_ + (size_t)r * H_ + c] = bf2f(hx[(size_t)r * H_ + c]);
  }
}

// ---------------------------------------------------------------------------
extern "C" void kernel_launch(void* const* d_in, const int* in_sizes, int n_in,
                              void* d_out, int out_size, void* d_ws, size_t ws_size,
                              hipStream_t stream) {
  (void)in_sizes; (void)n_in; (void)out_size; (void)ws_size;

  const float* x     = (const float*)d_in[0];
  const float* Wi_f  = (const float*)d_in[1];
  const float* bi    = (const float*)d_in[2];
  const float* Wh_f  = (const float*)d_in[3];
  const float* bbW_f = (const float*)d_in[4];
  const float* bbb   = (const float*)d_in[5];
  const float* ff1W  = (const float*)d_in[6];
  const float* ff1b  = (const float*)d_in[7];
  const float* ff2W  = (const float*)d_in[8];
  const float* ff2b  = (const float*)d_in[9];
  const float* taW   = (const float*)d_in[10];
  const float* tab   = (const float*)d_in[11];
  const float* tbW   = (const float*)d_in[12];
  const float* tbb   = (const float*)d_in[13];
  const float* hW    = (const float*)d_in[14];
  const float* hb    = (const float*)d_in[15];

  // ws layout: sync(256B) | Wi | Wh | Wbb | Wff(4) | hx | hlx | fx | hpart | xbf
  char* ws = (char*)d_ws;
  unsigned* syncp = (unsigned*)ws;
  __bf16* Wi   = (__bf16*)(ws + 256);
  __bf16* Wh   = Wi  + (size_t)G4H * I_;
  __bf16* Wbb  = Wh  + (size_t)G4H * H_;
  __bf16* Wff  = Wbb + (size_t)BB_ * (I_ + H_);
  __bf16* hx   = Wff + (size_t)4 * H_ * BB_;
  __bf16* hlx  = hx  + (size_t)B_ * H_;
  __bf16* fx   = hlx + (size_t)B_ * H_;
  float*  hpart = (float*)(fx + (size_t)B_ * BB_);
  __bf16* xbf  = (__bf16*)(hpart + (size_t)NWG * B_ * O_);

  // per-launch reset (graph-replay safe): barrier counter + h0 = 0
  (void)hipMemsetAsync(syncp, 0, 256, stream);
  (void)hipMemsetAsync(hx, 0, (size_t)B_ * H_ * 2, stream);

  // bf16 conversions (vectorized x4)
  auto cvt = [&](const float* s, __bf16* d, size_t n) {
    int n4 = (int)(n / 4);
    cvt_bf16x4<<<(n4 + 255) / 256, 256, 0, stream>>>((const float4*)s, (uint2*)d, n4);
  };
  cvt(Wi_f,  Wi,  (size_t)G4H * I_);
  cvt(Wh_f,  Wh,  (size_t)G4H * H_);
  cvt(bbW_f, Wbb, (size_t)BB_ * (I_ + H_));
  cvt(ff1W,  Wff + 0 * (size_t)H_ * BB_, (size_t)H_ * BB_);
  cvt(ff2W,  Wff + 1 * (size_t)H_ * BB_, (size_t)H_ * BB_);
  cvt(taW,   Wff + 2 * (size_t)H_ * BB_, (size_t)H_ * BB_);
  cvt(tbW,   Wff + 3 * (size_t)H_ * BB_, (size_t)H_ * BB_);
  cvt(x,     xbf, (size_t)B_ * T_ * I_);

  (void)hipFuncSetAttribute((const void*)cfc_scan_kernel,
                            hipFuncAttributeMaxDynamicSharedMemorySize, DYN_LDS);

  cfc_scan_kernel<<<dim3(NWG), dim3(256), DYN_LDS, stream>>>(
      xbf, bi, bbb, ff1b, ff2b, tab, tbb, hW, hb,
      Wi, Wh, Wbb, Wff, hx, hlx, fx, hpart, syncp, (float*)d_out);
}
